// RegressionFusionOutput_9758165697301
// MI455X (gfx1250) — matrix operand order repaired and checked by reading, not compiled, not run on hardware
//
#include <hip/hip_runtime.h>
#include <hip/hip_bf16.h>

typedef __attribute__((ext_vector_type(16))) __bf16 v16bf;
typedef __attribute__((ext_vector_type(8)))  float  v8f;
typedef unsigned short u16t;

#define N_    2048
#define R_    36
#define D_    1024
#define I_    2048
#define MTOT  (N_ * R_)      // 73728 rows
#define MTILES (MTOT / 32)   // 2304 workgroups, 32 rows each

__device__ __forceinline__ u16t f2bf(float f) {
    unsigned u = __float_as_uint(f);
    u += 0x7fffu + ((u >> 16) & 1u);   // round-to-nearest-even
    return (u16t)(u >> 16);
}
__device__ __forceinline__ float bf2f(u16t h) {
    return __uint_as_float(((unsigned)h) << 16);
}
__device__ __forceinline__ float wred32(float v) {
#pragma unroll
    for (int m = 16; m >= 1; m >>= 1) v += __shfl_xor(v, m, 32);
    return v;
}
__device__ __forceinline__ float wred16(float v) {   // reduce within 16-lane half
#pragma unroll
    for (int m = 8; m >= 1; m >>= 1) v += __shfl_xor(v, m, 32);
    return v;
}
__device__ __forceinline__ v8f wmma_bf16(v16bf a, v16bf b, v8f c) {
    return __builtin_amdgcn_wmma_f32_16x16x32_bf16(false, a, false, b, (short)0, c,
                                                   false, false);
}

// ---------------- kernel 1: q = mean over R of q_emb ----------------
__global__ __launch_bounds__(256) void qmean_kernel(const float* __restrict__ q_emb,
                                                    float* __restrict__ qmean) {
    int idx = blockIdx.x * 256 + threadIdx.x;      // N_*D_ threads
    int n = idx >> 10, d = idx & 1023;
    const float* p = q_emb + (size_t)n * R_ * D_ + d;
    float s = 0.f;
#pragma unroll
    for (int r = 0; r < R_; ++r) s += p[(size_t)r * D_];
    qmean[idx] = s * (1.0f / R_);
}

// ---- kernel 2: split w1 into bf16 hi/lo, store in WMMA B-fragment order ----
// B frag (32x16, 16-bit): lane l = khalf*16 + n ; elem j: k5 = (j<8 ? khalf*8+j : 16+khalf*8+j-8)
__global__ __launch_bounds__(256) void w1prep_kernel(const float* __restrict__ w1,
                                                     u16t* __restrict__ w1h,
                                                     u16t* __restrict__ w1l) {
    int idx = blockIdx.x * 256 + threadIdx.x;      // D_*I_ threads
    int k = idx >> 11, i = idx & 2047;
    float w = w1[(size_t)k * I_ + i];
    u16t hi = f2bf(w);
    u16t lo = f2bf(w - bf2f(hi));
    int ks = k >> 5, k5 = k & 31, c = i >> 4, n = i & 15;
    int l = ((k5 >> 3) & 1) * 16 + n, j = (k5 & 7) | (((k5 >> 4) & 1) << 3);
    size_t dst = (((size_t)ks * 128 + c) * 32 + l) * 16 + j;
    w1h[dst] = hi;
    w1l[dst] = lo;
}

// ---------------- kernel 3: zero pred (harness does not re-poison) ----------
__global__ __launch_bounds__(256) void zero_pred_kernel(float* __restrict__ pred) {
    pred[blockIdx.x * 256 + threadIdx.x] = 0.f;
}

// ------ kernel 4: fused LN1 -> GEMM(bf16x3, 32-row tile) -> GELU -> LN2 -> w2
__global__ __launch_bounds__(512) void head_kernel(
    const float* __restrict__ v_emb, const float* __restrict__ qmean,
    const float* __restrict__ ln1g, const float* __restrict__ ln1b,
    const u16t* __restrict__ w1h, const u16t* __restrict__ w1l,
    const float* __restrict__ b1, const float* __restrict__ ln2g,
    const float* __restrict__ ln2b, const float* __restrict__ w2,
    const float* __restrict__ b2, float* __restrict__ scores,
    float* __restrict__ pred) {
    // Two 16-row A tiles (rowtile t = 0,1), each 16 x 1024 K, bf16 hi/lo,
    // stored in WMMA A-fragment layout:
    //   frag(ks): lane l = khalf*16 + m ; elem j: k5 = (j<8 ? khalf*8+j : 16+khalf*8+j-8)
    __shared__ alignas(32) u16t Ah[2 * 32 * 32 * 16];  // 64 KB
    __shared__ alignas(32) u16t Al[2 * 32 * 32 * 16];  // 64 KB

    const int tid = threadIdx.x;
    const int wave = tid >> 5, lane = tid & 31;   // 16 waves
    const int mtile = blockIdx.x;                 // 32 rows per block

    // ---- prologue: fusion + LayerNorm1, split to bf16 hi/lo, swizzled LDS store
#pragma unroll
    for (int rr = 0; rr < 2; ++rr) {
        const int rloc = wave * 2 + rr;              // 0..31
        const int row = mtile * 32 + rloc;           // row == n*R_ + r (contiguous)
        const int n = row / R_;
        const float* vp = v_emb + (size_t)row * D_;
        const float* qp = qmean + (size_t)n * D_;
        float x[32];
        float s = 0.f;
#pragma unroll
        for (int i = 0; i < 32; ++i) {
            int d = i * 32 + lane;
            x[i] = vp[d] * qp[d];
            s += x[i];
        }
        s = wred32(s);
        const float mu = s * (1.0f / D_);
        float s2 = 0.f;
#pragma unroll
        for (int i = 0; i < 32; ++i) {
            float xc = x[i] - mu;
            s2 += xc * xc;
        }
        s2 = wred32(s2);
        const float rstd = rsqrtf(s2 * (1.0f / D_) + 1e-5f);
        const int t = rloc >> 4, m = rloc & 15;
        u16t* AhT = Ah + t * (32 * 32 * 16);
        u16t* AlT = Al + t * (32 * 32 * 16);
#pragma unroll
        for (int i = 0; i < 32; ++i) {
            int d = i * 32 + lane;
            float a = (x[i] - mu) * rstd * ln1g[d] + ln1b[d];
            u16t hi = f2bf(a);
            u16t lo = f2bf(a - bf2f(hi));
            int ks = d >> 5, k5 = d & 31;
            int khalf = (k5 >> 3) & 1;
            int j = (k5 & 7) | (((k5 >> 4) & 1) << 3);
            int addr = ((ks * 32) + (khalf * 16 + m)) * 16 + j;
            AhT[addr] = hi;
            AlT[addr] = lo;
        }
    }
    __syncthreads();

    // ---- main GEMM: 32 x 2048, K = 1024; wave owns 128 columns (8 col-tiles)
    //      and BOTH row-tiles, so every B fragment feeds 6 WMMAs.
    v8f acc0[8] = {};   // rowtile 0
    v8f acc1[8] = {};   // rowtile 1
    const size_t wcol = (size_t)wave * 8;   // col-tile base for this wave
    for (int ks = 0; ks < 32; ++ks) {
        const int abase = (ks * 32 + lane) * 16;
        const v16bf ah0 = *(const v16bf*)&Ah[abase];
        const v16bf al0 = *(const v16bf*)&Al[abase];
        const v16bf ah1 = *(const v16bf*)&Ah[32 * 32 * 16 + abase];
        const v16bf al1 = *(const v16bf*)&Al[32 * 32 * 16 + abase];
        const u16t* bhp = w1h + (((size_t)ks * 128 + wcol) * 32 + lane) * 16;
        const u16t* blp = w1l + (((size_t)ks * 128 + wcol) * 32 + lane) * 16;
#pragma unroll
        for (int c = 0; c < 8; ++c) {
            v16bf bh = *(const v16bf*)(bhp + (size_t)c * 512);
            v16bf bl = *(const v16bf*)(blp + (size_t)c * 512);
            acc0[c] = wmma_bf16(ah0, bh, acc0[c]);
            acc0[c] = wmma_bf16(ah0, bl, acc0[c]);
            acc0[c] = wmma_bf16(al0, bh, acc0[c]);
            acc1[c] = wmma_bf16(ah1, bh, acc1[c]);
            acc1[c] = wmma_bf16(ah1, bl, acc1[c]);
            acc1[c] = wmma_bf16(al1, bh, acc1[c]);
        }
    }

    // ---- epilogue. C layout: lane l, vgpr v -> column n = l&15, row m = (l>>4)*8+v
    const int nIdx = lane & 15;
    const int half = lane >> 4;
    float b1v[8];
#pragma unroll
    for (int c = 0; c < 8; ++c) b1v[c] = b1[wave * 128 + c * 16 + nIdx];

    // GELU + per-row partial sums (both row-tiles)
    float sum0[8] = {}, sq0[8] = {}, sum1[8] = {}, sq1[8] = {};
#pragma unroll
    for (int c = 0; c < 8; ++c) {
#pragma unroll
        for (int v = 0; v < 8; ++v) {
            float y0 = acc0[c][v] + b1v[c];
            float g0 = 0.5f * y0 * (1.0f + erff(y0 * 0.70710678118654752f));
            acc0[c][v] = g0;
            sum0[v] += g0;
            sq0[v] += g0 * g0;
            float y1 = acc1[c][v] + b1v[c];
            float g1 = 0.5f * y1 * (1.0f + erff(y1 * 0.70710678118654752f));
            acc1[c][v] = g1;
            sum1[v] += g1;
            sq1[v] += g1 * g1;
        }
    }
#pragma unroll
    for (int v = 0; v < 8; ++v) {
        sum0[v] = wred16(sum0[v]);
        sq0[v] = wred16(sq0[v]);
        sum1[v] = wred16(sum1[v]);
        sq1[v] = wred16(sq1[v]);
    }

    __syncthreads();                 // A tiles dead: reuse LDS as reduction scratch
    float* red = (float*)Ah;
    // layout (floats): [0,512) sums, [512,1024) sq, [1024,1056) mu,
    //                  [1056,1088) rstd, [1088,1600) dot
    if (nIdx == 0) {
#pragma unroll
        for (int v = 0; v < 8; ++v) {
            red[wave * 32 + half * 8 + v] = sum0[v];
            red[wave * 32 + 16 + half * 8 + v] = sum1[v];
            red[512 + wave * 32 + half * 8 + v] = sq0[v];
            red[512 + wave * 32 + 16 + half * 8 + v] = sq1[v];
        }
    }
    __syncthreads();
    if (tid < 32) {
        float s = 0.f, s2 = 0.f;
#pragma unroll
        for (int w = 0; w < 16; ++w) {
            s += red[w * 32 + tid];
            s2 += red[512 + w * 32 + tid];
        }
        float mu = s * (1.0f / I_);
        float var = s2 * (1.0f / I_) - mu * mu;
        red[1024 + tid] = mu;
        red[1056 + tid] = rsqrtf(var + 1e-5f);
    }
    __syncthreads();

    float g2v[8], bb2v[8], w2v[8];
#pragma unroll
    for (int c = 0; c < 8; ++c) {
        int col = wave * 128 + c * 16 + nIdx;
        g2v[c] = ln2g[col];
        bb2v[c] = ln2b[col];
        w2v[c] = w2[col];
    }
    float dot0[8] = {}, dot1[8] = {};
#pragma unroll
    for (int v = 0; v < 8; ++v) {
        float mu0 = red[1024 + half * 8 + v], rs0 = red[1056 + half * 8 + v];
        float mu1 = red[1024 + 16 + half * 8 + v], rs1 = red[1056 + 16 + half * 8 + v];
#pragma unroll
        for (int c = 0; c < 8; ++c) {
            float z0 = (acc0[c][v] - mu0) * rs0 * g2v[c] + bb2v[c];
            dot0[v] += z0 * w2v[c];
            float z1 = (acc1[c][v] - mu1) * rs1 * g2v[c] + bb2v[c];
            dot1[v] += z1 * w2v[c];
        }
    }
#pragma unroll
    for (int v = 0; v < 8; ++v) {
        dot0[v] = wred16(dot0[v]);
        dot1[v] = wred16(dot1[v]);
    }
    if (nIdx == 0) {
#pragma unroll
        for (int v = 0; v < 8; ++v) {
            red[1088 + wave * 32 + half * 8 + v] = dot0[v];
            red[1088 + wave * 32 + 16 + half * 8 + v] = dot1[v];
        }
    }
    __syncthreads();
    if (tid < 32) {
        float d = 0.f;
#pragma unroll
        for (int w = 0; w < 16; ++w) d += red[1088 + w * 32 + tid];
        float sc = 1.0f / (1.0f + expf(-(d + b2[0])));
        int row = mtile * 32 + tid;
        scores[row] = sc;
        atomicAdd(&pred[row / R_], sc);
    }
}

// ---------------- kernel 5: one-hot logits -----------------------------------
__global__ __launch_bounds__(256) void logits_kernel(const float* __restrict__ pred,
                                                     float* __restrict__ logits) {
    int n = blockIdx.x * 256 + threadIdx.x;
    float p = pred[n];
    int aid = (int)rintf(p);               // round half-to-even (matches torch)
    aid = aid < 0 ? 0 : (aid > 15 ? 15 : aid);
#pragma unroll
    for (int j = 0; j < 16; ++j) logits[n * 16 + j] = (j == aid) ? 1.0f : 0.0f;
}

extern "C" void kernel_launch(void* const* d_in, const int* in_sizes, int n_in,
                              void* d_out, int out_size, void* d_ws, size_t ws_size,
                              hipStream_t stream) {
    (void)in_sizes; (void)n_in; (void)out_size; (void)ws_size;
    const float* v_emb = (const float*)d_in[0];
    const float* q_emb = (const float*)d_in[1];
    const float* ln1g  = (const float*)d_in[2];
    const float* ln1b  = (const float*)d_in[3];
    const float* w1    = (const float*)d_in[4];
    const float* b1    = (const float*)d_in[5];
    const float* ln2g  = (const float*)d_in[6];
    const float* ln2b  = (const float*)d_in[7];
    const float* w2    = (const float*)d_in[8];
    const float* b2    = (const float*)d_in[9];

    float* out    = (float*)d_out;
    float* scores = out;                 // [73728]
    float* pred   = out + MTOT;          // [2048]
    float* logits = pred + N_;           // [32768]

    char* ws = (char*)d_ws;
    float* qmean = (float*)ws;                                   // 8 MB
    u16t*  w1h   = (u16t*)(ws + (size_t)8  * 1024 * 1024);       // 4 MB
    u16t*  w1l   = (u16t*)(ws + (size_t)12 * 1024 * 1024);       // 4 MB

    qmean_kernel<<<(N_ * D_) / 256, 256, 0, stream>>>(q_emb, qmean);
    w1prep_kernel<<<(D_ * I_) / 256, 256, 0, stream>>>(w1, w1h, w1l);
    zero_pred_kernel<<<N_ / 256, 256, 0, stream>>>(pred);
    head_kernel<<<MTILES, 512, 0, stream>>>(v_emb, qmean, ln1g, ln1b, w1h, w1l, b1,
                                            ln2g, ln2b, w2, b2, scores, pred);
    logits_kernel<<<N_ / 256, 256, 0, stream>>>(pred, logits);
}
